// PointNetDecoder_7301444403788
// MI455X (gfx1250) — matrix operand order repaired and checked by reading, not compiled, not run on hardware
//
#include <hip/hip_runtime.h>
#include <float.h>

// ---------------------------------------------------------------------------
// CDNA5 (gfx1250) WMMA types
// ---------------------------------------------------------------------------
typedef __attribute__((ext_vector_type(16))) __bf16         v16bf;
typedef __attribute__((ext_vector_type(8)))  float          v8f;
typedef __attribute__((ext_vector_type(8)))  unsigned short v8u16;
typedef __attribute__((ext_vector_type(16))) unsigned short v16u16;
typedef unsigned short u16t;

__device__ __forceinline__ u16t f2bfu(float f) {
  // round-to-nearest-even f32 -> bf16 bits
  union { float f; unsigned u; } x; x.f = f;
  unsigned r = x.u + 0x7FFFu + ((x.u >> 16) & 1u);
  return (u16t)(r >> 16);
}

__device__ __forceinline__ v16bf cat_bf(v8u16 lo, v8u16 hi) {
  return __builtin_bit_cast(v16bf, __builtin_shufflevector(
      lo, hi, 0, 1, 2, 3, 4, 5, 6, 7, 8, 9, 10, 11, 12, 13, 14, 15));
}

// ---------------------------------------------------------------------------
// kNN interpolate (k=3): one wave32 per query point.
// ---------------------------------------------------------------------------
__device__ __forceinline__ void ins3(float d, int i, float* bd, int* bi) {
  if (d < bd[2]) {
    if (d < bd[0]) {
      bd[2] = bd[1]; bi[2] = bi[1];
      bd[1] = bd[0]; bi[1] = bi[0];
      bd[0] = d;     bi[0] = i;
    } else if (d < bd[1]) {
      bd[2] = bd[1]; bi[2] = bi[1];
      bd[1] = d;     bi[1] = i;
    } else {
      bd[2] = d;     bi[2] = i;
    }
  }
}

__global__ void knn_interp_kernel(const float* __restrict__ X,   // [Ns, C]
                                  const float* __restrict__ P,   // [Ns, 3]
                                  const float* __restrict__ Q,   // [Nq, 3]
                                  float* __restrict__ Out,       // [Nq, C]
                                  int Ns, int Nq, int C) {
  const int lane = threadIdx.x & 31;
  const int wave = threadIdx.x >> 5;
  const int q = blockIdx.x * (blockDim.x >> 5) + wave;
  if (q >= Nq) return;

  const float qx = Q[q * 3 + 0];
  const float qy = Q[q * 3 + 1];
  const float qz = Q[q * 3 + 2];

  float bd[3] = {FLT_MAX, FLT_MAX, FLT_MAX};
  int   bi[3] = {0, 0, 0};

  for (int s = lane; s < Ns; s += 32) {
    __builtin_prefetch(&P[3 * (s + 256)], 0, 0);   // global_prefetch_b8
    float dx = qx - P[s * 3 + 0];
    float dy = qy - P[s * 3 + 1];
    float dz = qz - P[s * 3 + 2];
    ins3(dx * dx + dy * dy + dz * dz, s, bd, bi);
  }

  // butterfly merge of per-lane sorted-3 lists (read remote BEFORE inserting)
  for (int off = 16; off >= 1; off >>= 1) {
    float od0 = __shfl_down(bd[0], off, 32);
    float od1 = __shfl_down(bd[1], off, 32);
    float od2 = __shfl_down(bd[2], off, 32);
    int   oi0 = __shfl_down(bi[0], off, 32);
    int   oi1 = __shfl_down(bi[1], off, 32);
    int   oi2 = __shfl_down(bi[2], off, 32);
    ins3(od0, oi0, bd, bi);
    ins3(od1, oi1, bd, bi);
    ins3(od2, oi2, bd, bi);
  }

  const float d0 = __shfl(bd[0], 0, 32);
  const float d1 = __shfl(bd[1], 0, 32);
  const float d2 = __shfl(bd[2], 0, 32);
  const int   i0 = __shfl(bi[0], 0, 32);
  const int   i1 = __shfl(bi[1], 0, 32);
  const int   i2 = __shfl(bi[2], 0, 32);

  const float w0 = 1.0f / fmaxf(d0, 1e-16f);
  const float w1 = 1.0f / fmaxf(d1, 1e-16f);
  const float w2 = 1.0f / fmaxf(d2, 1e-16f);
  const float inv = 1.0f / (w0 + w1 + w2);

  const float* f0 = X + (size_t)i0 * C;
  const float* f1 = X + (size_t)i1 * C;
  const float* f2 = X + (size_t)i2 * C;
  float* o = Out + (size_t)q * C;

  if ((C & 127) == 0) {
    // wide path: float4 (b128) loads/stores, 128 channels per sweep
    for (int c = lane * 4; c < C; c += 128) {
      const float4 a = *(const float4*)(f0 + c);
      const float4 b = *(const float4*)(f1 + c);
      const float4 d = *(const float4*)(f2 + c);
      float4 r;
      r.x = (w0 * a.x + w1 * b.x + w2 * d.x) * inv;
      r.y = (w0 * a.y + w1 * b.y + w2 * d.y) * inv;
      r.z = (w0 * a.z + w1 * b.z + w2 * d.z) * inv;
      r.w = (w0 * a.w + w1 * b.w + w2 * d.w) * inv;
      *(float4*)(o + c) = r;
    }
  } else {
    for (int c = lane; c < C; c += 32)
      o[c] = (w0 * f0[c] + w1 * f1[c] + w2 * f2[c]) * inv;
  }
}

// ---------------------------------------------------------------------------
// pack_a: Apk[M, Kp] bf16 = zero-padded concat(A1[M,C1], A2[M,C2]).
// One thread per K-pair -> one 32-bit store.
// ---------------------------------------------------------------------------
__global__ void pack_a_kernel(const float* __restrict__ A1, int C1,
                              const float* __restrict__ A2, int C2,
                              int Kt, int Kp, int M,
                              unsigned int* __restrict__ Apk) { // [M, Kp/2]
  const int halfKp = Kp >> 1;
  const long long total = (long long)M * halfKp;
  long long idx = (long long)blockIdx.x * blockDim.x + threadIdx.x;
  if (idx >= total) return;
  const int row = (int)(idx / halfKp);
  const int k   = ((int)(idx % halfKp)) << 1;

  float v0 = 0.f, v1 = 0.f;
  if (k < Kt)
    v0 = (k < C1) ? A1[(size_t)row * C1 + k] : A2[(size_t)row * C2 + (k - C1)];
  if (k + 1 < Kt)
    v1 = (k + 1 < C1) ? A1[(size_t)row * C1 + k + 1]
                      : A2[(size_t)row * C2 + (k + 1 - C1)];
  Apk[idx] = (unsigned)f2bfu(v0) | ((unsigned)f2bfu(v1) << 16);
}

// ---------------------------------------------------------------------------
// pack_w: repack W[Kt, N] (row-major f32) into WMMA B-operand tile order.
// Layout: Wpk[nt][kc][lane][16 bf16]; lane's elements 0..7 are K(kc*32 +
// laneHi*8 + 0..7), 8..15 are K(kc*32 + 16 + laneHi*8 + 0..7) of column nt*16 + lane&15.
// ---------------------------------------------------------------------------
__global__ void pack_w_kernel(const float* __restrict__ W, int Kt, int N,
                              int KC, int TN, u16t* __restrict__ Wpk) {
  const int total = TN * KC * 32;
  const int tid = blockIdx.x * blockDim.x + threadIdx.x;
  if (tid >= total) return;
  const int lane = tid & 31;
  const int kci  = (tid >> 5) % KC;
  const int nt   = (tid >> 5) / KC;
  const int n    = (nt << 4) + (lane & 15);
  const int kb   = (kci << 5) + ((lane >> 4) << 3);

  u16t vals[16];
#pragma unroll
  for (int j = 0; j < 16; ++j) {
    const int k = kb + j + ((j >> 3) << 3);
    float v = (k < Kt && n < N) ? W[(size_t)k * N + n] : 0.f;
    vals[j] = f2bfu(v);
  }
  v16u16 pk;
#pragma unroll
  for (int j = 0; j < 16; ++j) pk[j] = vals[j];
  *(v16u16*)(Wpk + (size_t)tid * 16) = pk;
}

// ---------------------------------------------------------------------------
// Branch-free WMMA GEMM on pre-packed bf16 operands, register-blocked over
// MB=4 M-tiles (64 rows) per wave, software-pipelined (register double-
// buffering): each iteration issues the 10 b128 loads for chunk k+1, then
// runs 4 WMMAs on chunk k's registers -> loads stay in flight across WMMAs.
// A layout (16x32 bf16): lane m=l&15; elements 0..7 = K(laneHi*8 + 0..7),
// elements 8..15 = K(16 + laneHi*8 + 0..7) -> two contiguous 16B runs.
// C/D layout (16x16 f32, 8 VGPRs): row = r + 8*laneHi, col = l&15.
// ---------------------------------------------------------------------------
__global__ void wmma_gemm_pk(const u16t* __restrict__ Apk,  // [M, Kp]
                             const u16t* __restrict__ Wpk,  // tile-packed
                             const float* __restrict__ bias,
                             float* __restrict__ Out,       // [M, N]
                             int M, int N, int Kp, int doRelu) {
  const int lane   = threadIdx.x & 31;
  const int wave   = threadIdx.x >> 5;
  const int tilesN = (N + 15) >> 4;
  const int tilesM = M >> 6;                // groups of 4 M-tiles (M % 64 == 0)
  const int tile   = blockIdx.x * (blockDim.x >> 5) + wave;
  if (tile >= tilesM * tilesN) return;      // wave-uniform: EXEC all-1 below

  const int tm     = (tile / tilesN) << 6;  // first of 64 rows
  const int nt     = tile % tilesN;
  const int laneHi = lane >> 4;
  const int lmod   = lane & 15;
  const int KC     = Kp >> 5;

  const u16t* __restrict__ arow =
      Apk + (size_t)(tm + lmod) * Kp + (laneHi << 3);
  const u16t* __restrict__ bptr = Wpk + ((size_t)nt * KC * 32 + lane) * 16;
  const size_t rstep = (size_t)16 * Kp;     // 16 rows of A

  v8f acc0 = {0.f, 0.f, 0.f, 0.f, 0.f, 0.f, 0.f, 0.f};
  v8f acc1 = acc0, acc2 = acc0, acc3 = acc0;

  // prologue: load chunk 0
  v16u16 bcur = *(const v16u16*)(bptr);
  v8u16 aLo[4], aHi[4];
#pragma unroll
  for (int i = 0; i < 4; ++i) {
    aLo[i] = *(const v8u16*)(arow + (size_t)i * rstep);
    aHi[i] = *(const v8u16*)(arow + (size_t)i * rstep + 16);
  }

  // steady state: prefetch chunk k+1, WMMA on chunk k
  for (int kci = 0; kci < KC - 1; ++kci) {
    arow += 32;
    bptr += 512;
    const v16u16 bnxt = *(const v16u16*)(bptr);
    v8u16 nLo[4], nHi[4];
#pragma unroll
    for (int i = 0; i < 4; ++i) {
      nLo[i] = *(const v8u16*)(arow + (size_t)i * rstep);
      nHi[i] = *(const v8u16*)(arow + (size_t)i * rstep + 16);
    }

    const v16bf b = __builtin_bit_cast(v16bf, bcur);
    acc0 = __builtin_amdgcn_wmma_f32_16x16x32_bf16(false, cat_bf(aLo[0], aHi[0]),
                                                   false, b, (short)0, acc0, false, false);
    acc1 = __builtin_amdgcn_wmma_f32_16x16x32_bf16(false, cat_bf(aLo[1], aHi[1]),
                                                   false, b, (short)0, acc1, false, false);
    acc2 = __builtin_amdgcn_wmma_f32_16x16x32_bf16(false, cat_bf(aLo[2], aHi[2]),
                                                   false, b, (short)0, acc2, false, false);
    acc3 = __builtin_amdgcn_wmma_f32_16x16x32_bf16(false, cat_bf(aLo[3], aHi[3]),
                                                   false, b, (short)0, acc3, false, false);

    bcur = bnxt;
#pragma unroll
    for (int i = 0; i < 4; ++i) { aLo[i] = nLo[i]; aHi[i] = nHi[i]; }
  }

  // epilogue chunk
  {
    const v16bf b = __builtin_bit_cast(v16bf, bcur);
    acc0 = __builtin_amdgcn_wmma_f32_16x16x32_bf16(false, cat_bf(aLo[0], aHi[0]),
                                                   false, b, (short)0, acc0, false, false);
    acc1 = __builtin_amdgcn_wmma_f32_16x16x32_bf16(false, cat_bf(aLo[1], aHi[1]),
                                                   false, b, (short)0, acc1, false, false);
    acc2 = __builtin_amdgcn_wmma_f32_16x16x32_bf16(false, cat_bf(aLo[2], aHi[2]),
                                                   false, b, (short)0, acc2, false, false);
    acc3 = __builtin_amdgcn_wmma_f32_16x16x32_bf16(false, cat_bf(aLo[3], aHi[3]),
                                                   false, b, (short)0, acc3, false, false);
  }

  const int col = (nt << 4) + lmod;
  if (col < N) {
    const float bv = bias[col];
    v8f accs[4] = {acc0, acc1, acc2, acc3};
#pragma unroll
    for (int i = 0; i < 4; ++i) {
      const int rbase = tm + (i << 4) + (laneHi << 3);
#pragma unroll
      for (int r = 0; r < 8; ++r) {
        float vo = accs[i][r] + bv;
        if (doRelu) vo = fmaxf(vo, 0.f);
        Out[(size_t)(rbase + r) * N + col] = vo;
      }
    }
  }
}

// ---------------------------------------------------------------------------
// Host-side orchestration
// ---------------------------------------------------------------------------
static inline void launch_knn(hipStream_t s, const float* X, const float* P,
                              const float* Q, float* O, int Ns, int Nq, int C) {
  dim3 blk(256), grd((Nq + 7) / 8);   // 8 waves/block, one query per wave
  hipLaunchKernelGGL(knn_interp_kernel, grd, blk, 0, s, X, P, Q, O, Ns, Nq, C);
}

static inline void launch_gemm(hipStream_t s,
                               const float* A1, int C1, const float* A2, int C2,
                               const float* W, const float* b, float* O,
                               int M, int N, int Kt, int relu,
                               u16t* Apk, u16t* Wpk) {
  const int Kp = (Kt + 31) & ~31;
  const int KC = Kp >> 5;
  const int TN = (N + 15) >> 4;

  {
    const long long total = (long long)M * (Kp >> 1);
    dim3 blk(256), grd((unsigned)((total + 255) / 256));
    hipLaunchKernelGGL(pack_a_kernel, grd, blk, 0, s,
                       A1, C1, A2, C2, Kt, Kp, M, (unsigned int*)Apk);
  }
  {
    const int total = TN * KC * 32;
    dim3 blk(256), grd((total + 255) / 256);
    hipLaunchKernelGGL(pack_w_kernel, grd, blk, 0, s, W, Kt, N, KC, TN, Wpk);
  }
  {
    const int tiles = (M >> 6) * TN;       // 4 M-tiles per wave (M % 64 == 0)
    dim3 blk(256), grd((tiles + 7) / 8);   // 8 waves per block
    hipLaunchKernelGGL(wmma_gemm_pk, grd, blk, 0, s,
                       Apk, Wpk, b, O, M, N, Kp, relu);
  }
}

extern "C" void kernel_launch(void* const* d_in, const int* in_sizes, int n_in,
                              void* d_out, int out_size, void* d_ws, size_t ws_size,
                              hipStream_t stream) {
  (void)in_sizes; (void)n_in; (void)out_size; (void)ws_size;

  const float* x0     = (const float*)d_in[0];
  const float* pos0   = (const float*)d_in[1];
  const float* x1     = (const float*)d_in[2];
  const float* pos1   = (const float*)d_in[3];
  const float* x2     = (const float*)d_in[4];
  const float* pos2   = (const float*)d_in[5];
  const float* x3     = (const float*)d_in[6];
  const float* pos3   = (const float*)d_in[7];
  // d_in[8..11] = batch vectors (all zero; single point cloud) — unused
  const float* fp3_W0 = (const float*)d_in[12];
  const float* fp3_b0 = (const float*)d_in[13];
  const float* fp3_W1 = (const float*)d_in[14];
  const float* fp3_b1 = (const float*)d_in[15];
  const float* fp2_W0 = (const float*)d_in[16];
  const float* fp2_b0 = (const float*)d_in[17];
  const float* fp2_W1 = (const float*)d_in[18];
  const float* fp2_b1 = (const float*)d_in[19];
  const float* fp1_W0 = (const float*)d_in[20];
  const float* fp1_b0 = (const float*)d_in[21];
  const float* fp1_W1 = (const float*)d_in[22];
  const float* fp1_b1 = (const float*)d_in[23];
  const float* fp1_W2 = (const float*)d_in[24];
  const float* fp1_b2 = (const float*)d_in[25];
  const float* lin1_W = (const float*)d_in[26];
  const float* lin1_b = (const float*)d_in[27];
  const float* lin2_W = (const float*)d_in[28];
  const float* lin2_b = (const float*)d_in[29];

  float* out = (float*)d_out;

  // Workspace layout (all 16B-aligned):
  //   A, B : f32 ping-pong activation buffers, 16384*128 each (8 MB each)
  //   Apk  : packed bf16 A-operand, max 16384*160 elems (5.24 MB)
  //   Wpk  : packed bf16 B-operand, max 1280*256 elems (0.66 MB)
  const size_t BUF = (size_t)16384 * 128;
  float* A   = (float*)d_ws;
  float* B   = A + BUF;
  u16t*  Apk = (u16t*)(B + BUF);
  u16t*  Wpk = Apk + (size_t)16384 * 160;

  // Stage fp3: interp(x3 -> pos2), MLP([1280,256,256])
  launch_knn (stream, x3, pos3, pos2, A, /*Ns=*/256, /*Nq=*/1024, /*C=*/1024);
  launch_gemm(stream, A, 1024, x2, 256, fp3_W0, fp3_b0, B, 1024, 256, 1280, 1, Apk, Wpk);
  launch_gemm(stream, B, 256,  B,  256, fp3_W1, fp3_b1, A, 1024, 256, 256, 1, Apk, Wpk);

  // Stage fp2: interp(-> pos1), MLP([384,256,128])
  launch_knn (stream, A, pos2, pos1, B, 1024, 4096, 256);
  launch_gemm(stream, B, 256, x1, 128, fp2_W0, fp2_b0, A, 4096, 256, 384, 1, Apk, Wpk);
  launch_gemm(stream, A, 256, A,  256, fp2_W1, fp2_b1, B, 4096, 128, 256, 1, Apk, Wpk);

  // Stage fp1: interp(-> pos0), MLP([138,128,128,128])
  launch_knn (stream, B, pos1, pos0, A, 4096, 16384, 128);
  launch_gemm(stream, A, 128, x0, 10, fp1_W0, fp1_b0, B, 16384, 128, 138, 1, Apk, Wpk);
  launch_gemm(stream, B, 128, B, 128, fp1_W1, fp1_b1, A, 16384, 128, 128, 1, Apk, Wpk);
  launch_gemm(stream, A, 128, A, 128, fp1_W2, fp1_b2, B, 16384, 128, 128, 1, Apk, Wpk);

  // Head: lin1 (ReLU) -> lin2 (no activation)
  launch_gemm(stream, B, 128, B, 128, lin1_W, lin1_b, A, 16384, 128, 128, 1, Apk, Wpk);
  launch_gemm(stream, A, 128, A, 128, lin2_W, lin2_b, out, 16384, 3, 128, 0, Apk, Wpk);
}
